// DSCDSCBlockLBPUNet_3745211482160
// MI455X (gfx1250) — matrix operand order repaired and checked by reading, not compiled, not run on hardware
//
#include <hip/hip_runtime.h>

// CDNA5 / gfx1250: wave32, WMMA bf16 16x16x32 (fp32 acc),
// branch-free async global->LDS staging, double-buffered tiles.
typedef __attribute__((ext_vector_type(16))) __bf16 v16bf;
typedef __attribute__((ext_vector_type(8)))  float  v8f;

#define C_IN    256
#define C_OUT   256
#define HW      128
#define BATCH   16
#define KCH     32                // input channels per K-chunk
#define NCHUNK  (C_IN / KCH)      // 8
#define MT      16                // pixel tile along W (GEMM M)
#define XSTRIDE 21                // 18 data cols + 3 trash cols (gcd(21,64)=1)

__global__ __launch_bounds__(256)
void convert_w1_bf16_kernel(const float* __restrict__ w1, __bf16* __restrict__ w1bf) {
    int i = blockIdx.x * 256 + threadIdx.x;     // 256 blocks x 256 thr = 65536
    w1bf[i] = (__bf16)w1[i];
}

template <bool WPRE>
__global__ __launch_bounds__(256)
void fused_bn_lbp_gemm_kernel(const float* __restrict__ x,
                              const float* __restrict__ gamma,
                              const float* __restrict__ beta,
                              const float* __restrict__ rmean,
                              const float* __restrict__ rvar,
                              const float* __restrict__ lbp,    // (C,1,3,3)
                              const float* __restrict__ w1,     // (O,C) fp32
                              const __bf16* __restrict__ w1bf,  // (O,C) bf16 (if WPRE)
                              const float* __restrict__ b1,
                              float* __restrict__ out)
{
    __shared__ float sInv[C_IN];
    __shared__ float sShift[C_IN];
    __shared__ float sWg[C_IN][9];
    __shared__ float sWsum[C_IN];
    // double-buffered raw x tile with halo: [buf][r*32+c][col], col 0..17 data
    __shared__ float sXr[2][3 * KCH][XSTRIDE];
    __shared__ __align__(16) __bf16 sA[MT][KCH];   // bf16 A tile (M x K)

    const int t  = threadIdx.x;
    const int wv = t >> 5;        // wave 0..7
    const int ln = t & 31;        // lane 0..31
    const int w0 = blockIdx.x * MT;
    const int h  = blockIdx.y;
    const int b  = blockIdx.z;

    // ---- one-time: BN params, ternary weights, weight sums, zero tiles ----
    {
        float iv = gamma[t] * rsqrtf(rvar[t] + 1e-5f);
        sInv[t]   = iv;
        sShift[t] = beta[t] - rmean[t] * iv;
        float ws = 0.0f;
        #pragma unroll
        for (int j = 0; j < 9; ++j) { float w = lbp[t * 9 + j]; sWg[t][j] = w; ws += w; }
        sWsum[t] = ws;
        float* z = &sXr[0][0][0];
        for (int e = t; e < 2 * 3 * KCH * XSTRIDE; e += 256) z[e] = 0.0f;
    }
    __syncthreads();   // zeros visible before any async copy lands

    // ---- loop-invariant staging geometry (wi == lane for all copies) ----
    const int  wi    = ln;
    const int  gw_s  = w0 + wi - 1;
    const bool gwOK  = (wi < 18) && ((unsigned)gw_s < (unsigned)HW);
    const int  gw_c  = gw_s < 0 ? 0 : (gw_s > HW - 1 ? HW - 1 : gw_s);
    const size_t xbase = (size_t)b * C_IN * HW * HW;
    const unsigned sXbase = (unsigned)(unsigned long long)&sXr[0][0][0];

    // WMMA fragment coordinates (wave32; A and B use contiguous k = kb + s)
    const int m_a  = ln & 15;           // A: row M
    const int kb_a = (ln >> 4) * 16;    // A: K base (0 or 16)
    const int n_f  = ln & 15;           // B/D: column N
    const int kb_b = (ln >> 4) * 16;    // B: K base (0 or 16)
    const int o0   = (wv * 2 + 0) * 16 + n_f;
    const int o1   = (wv * 2 + 1) * 16 + n_f;

    // branch-free async stage of one K-chunk into buffer bufi
    auto stage = [&](int kc, int bufi) {
        #pragma unroll
        for (int r = 0; r < 3; ++r) {
            int  gh   = h + r - 1;
            bool ghOK = (unsigned)gh < (unsigned)HW;       // wave-uniform
            int  gh_c = gh < 0 ? 0 : (gh > HW - 1 ? HW - 1 : gh);
            bool inb  = ghOK && gwOK;
            int  col  = inb ? wi : (18 + (ln & 1));        // OOB -> trash col
            #pragma unroll
            for (int j = 0; j < 4; ++j) {
                int c  = wv + (j << 3);                    // 0..31
                int cg = (kc << 5) + c;
                size_t idx = xbase + ((size_t)cg << 14) + (gh_c << 7) + gw_c;
                unsigned lds_off = sXbase +
                    (((unsigned)(bufi * 3 * KCH + (r << 5) + c) * XSTRIDE + col) << 2);
                unsigned long long ga = (unsigned long long)(x + idx);
                asm volatile("global_load_async_to_lds_b32 %0, %1, off"
                             :: "v"(lds_off), "v"(ga) : "memory");
            }
        }
    };

    v8f acc0 = {};
    v8f acc1 = {};

    stage(0, 0);   // prologue: chunk 0 in flight

    for (int kc = 0; kc < NCHUNK; ++kc) {
        const int cur = kc & 1;
        asm volatile("s_wait_asynccnt 0x0" ::: "memory");
        __syncthreads();   // tile[cur] landed; sA from kc-1 consumed; buf[cur^1] free

        // ---- depthwise 3x3 (BN folded) + ReLU + residual -> bf16 A tile ----
        {
            const float (*tile)[XSTRIDE] = sXr[cur];
            int c  = ln;
            int cg = (kc << 5) + c;
            float inv = sInv[cg];
            float sh  = sShift[cg];
            float wgt[9];
            #pragma unroll
            for (int j = 0; j < 9; ++j) wgt[j] = sWg[cg][j];
            int mb = wv * 2;
            #pragma unroll
            for (int mi = 0; mi < 2; ++mi) {
                int m = mb + mi;
                float S = 0.0f;                 // sum w * x_raw (OOB raw = 0)
                #pragma unroll
                for (int r = 0; r < 3; ++r)
                    #pragma unroll
                    for (int d = 0; d < 3; ++d)
                        S += tile[(r << 5) + c][m + d] * wgt[r * 3 + d];
                int gw = w0 + m;                // wave-uniform
                float Win;
                if (h > 0 && h < HW - 1 && gw > 0 && gw < HW - 1) {
                    Win = sWsum[cg];
                } else {
                    Win = 0.0f;
                    #pragma unroll
                    for (int r = 0; r < 3; ++r) {
                        bool rok = (unsigned)(h + r - 1) < (unsigned)HW;
                        #pragma unroll
                        for (int d = 0; d < 3; ++d) {
                            bool cok = (unsigned)(gw + d - 1) < (unsigned)HW;
                            Win += (rok && cok) ? wgt[r * 3 + d] : 0.0f;
                        }
                    }
                }
                float conv = inv * S + sh * Win;
                float xr   = tile[(1 << 5) + c][m + 1];    // raw center pixel
                sA[m][c] = (__bf16)(fmaxf(conv, 0.0f) + xr);
            }
        }

        // ---- prefetch next chunk under the WMMA (other buffer) ----
        if (kc + 1 < NCHUNK) stage(kc + 1, cur ^ 1);

        __syncthreads();   // sA ready for all waves

        // ---- A fragment from LDS (two 16B chunks, contiguous k = kb_a..kb_a+15) ----
        union { v16bf v; uint4 q[2]; } af;
        af.q[0] = *reinterpret_cast<const uint4*>(&sA[m_a][kb_a]);
        af.q[1] = *reinterpret_cast<const uint4*>(&sA[m_a][kb_a + 8]);

        // ---- B fragments ----
        union { v16bf v; uint4 q[2]; __bf16 e[16]; } bf0, bf1;
        if (WPRE) {
            const __bf16* p0 = w1bf + (size_t)o0 * C_IN + (kc << 5) + kb_b;
            const __bf16* p1 = w1bf + (size_t)o1 * C_IN + (kc << 5) + kb_b;
            bf0.q[0] = *reinterpret_cast<const uint4*>(p0);
            bf0.q[1] = *reinterpret_cast<const uint4*>(p0 + 8);
            bf1.q[0] = *reinterpret_cast<const uint4*>(p1);
            bf1.q[1] = *reinterpret_cast<const uint4*>(p1 + 8);
        } else {
            const float* wr0 = w1 + (size_t)o0 * C_IN + (kc << 5) + kb_b;
            const float* wr1 = w1 + (size_t)o1 * C_IN + (kc << 5) + kb_b;
            #pragma unroll
            for (int i = 0; i < 16; ++i) {
                bf0.e[i] = (__bf16)wr0[i];
                bf1.e[i] = (__bf16)wr1[i];
            }
        }

        acc0 = __builtin_amdgcn_wmma_f32_16x16x32_bf16(
                   false, af.v, false, bf0.v, (short)0, acc0, false, false);
        acc1 = __builtin_amdgcn_wmma_f32_16x16x32_bf16(
                   false, af.v, false, bf1.v, (short)0, acc1, false, false);
    }

    // ---- epilogue: bias + contiguous 2x b128 stores per lane per N-tile ----
    const int mb_out = (ln >> 4) * 8;   // D: lanes 0-15 -> M=r, 16-31 -> M=8+r
    {
        float bias = b1[o0];
        union { v8f v; float4 q[2]; } st;
        #pragma unroll
        for (int r = 0; r < 8; ++r) st.v[r] = acc0[r] + bias;
        size_t oidx = (((size_t)b * C_OUT + o0) * HW + h) * HW + (w0 + mb_out);
        *reinterpret_cast<float4*>(out + oidx)     = st.q[0];
        *reinterpret_cast<float4*>(out + oidx + 4) = st.q[1];
    }
    {
        float bias = b1[o1];
        union { v8f v; float4 q[2]; } st;
        #pragma unroll
        for (int r = 0; r < 8; ++r) st.v[r] = acc1[r] + bias;
        size_t oidx = (((size_t)b * C_OUT + o1) * HW + h) * HW + (w0 + mb_out);
        *reinterpret_cast<float4*>(out + oidx)     = st.q[0];
        *reinterpret_cast<float4*>(out + oidx + 4) = st.q[1];
    }
}

extern "C" void kernel_launch(void* const* d_in, const int* in_sizes, int n_in,
                              void* d_out, int out_size, void* d_ws, size_t ws_size,
                              hipStream_t stream) {
    const float* x     = (const float*)d_in[0];
    const float* gamma = (const float*)d_in[1];
    const float* beta  = (const float*)d_in[2];
    const float* rmean = (const float*)d_in[3];
    const float* rvar  = (const float*)d_in[4];
    const float* lbp   = (const float*)d_in[5];
    const float* w1    = (const float*)d_in[6];
    const float* b1    = (const float*)d_in[7];
    float* out = (float*)d_out;

    dim3 grid(HW / MT, HW, BATCH);   // (8, 128, 16) = 16384 workgroups
    const size_t w1bf_bytes = (size_t)C_OUT * C_IN * sizeof(unsigned short);
    if (ws_size >= w1bf_bytes) {
        __bf16* w1bf = (__bf16*)d_ws;
        convert_w1_bf16_kernel<<<(C_OUT * C_IN) / 256, 256, 0, stream>>>(w1, w1bf);
        fused_bn_lbp_gemm_kernel<true><<<grid, 256, 0, stream>>>(
            x, gamma, beta, rmean, rvar, lbp, w1, w1bf, b1, out);
    } else {
        fused_bn_lbp_gemm_kernel<false><<<grid, 256, 0, stream>>>(
            x, gamma, beta, rmean, rvar, lbp, w1, (const __bf16*)nullptr, b1, out);
    }
}
